// BitLinear158_48163763257939
// MI455X (gfx1250) — compile-verified, matrix operand order checked by reading, not run on hardware
//
#include <hip/hip_runtime.h>

// ---------------------------------------------------------------------------
// BitLinear 1.58b: Y[b,s,o] = sum_k x[b,s,k] * Wq[o,k] + bias[o]
//   Wq = clamp(round(W / (mean|W| + 1e-5)), -1, 1)  in {-1,0,1}
// x=[4,2048,4096] f32 -> M=8192, K=4096; W=[4096,4096] f32 [out,in] -> N=4096.
// f16 WMMA (ternary weights exact in f16), f32 accumulation.
// Wave tile 64x64 (16 WMMA : 16 ds_load_b128 per K-step), LDS double-buffered
// so staging of tile k+1 overlaps WMMA of tile k (1 barrier per K-step).
// ---------------------------------------------------------------------------

typedef _Float16 h8   __attribute__((ext_vector_type(8)));
typedef _Float16 v16h __attribute__((ext_vector_type(16)));
typedef float    v8f  __attribute__((ext_vector_type(8)));
typedef float    fv4  __attribute__((ext_vector_type(4)));

constexpr int MDIM = 8192;   // 4 * 2048
constexpr int NDIM = 4096;   // out features
constexpr int KDIM = 4096;   // in features
constexpr int BM = 256, BN = 128, BK = 32;
constexpr int LDT = BK + 8;          // LDS row stride in halves (80B, 16B-aligned)
constexpr int RED_BLOCKS = 256;
constexpr float QEPS = 1e-5f;

// ---------------- pass 1: partial sums of |W| (deterministic tree) ----------
__global__ __launch_bounds__(256)
void bl_abs_partial(const float* __restrict__ W, float* __restrict__ part, int n4) {
    __shared__ float red[256];
    const int tid = threadIdx.x;
    const fv4* W4 = (const fv4*)W;
    float s = 0.f;
    for (int i = blockIdx.x * blockDim.x + tid; i < n4; i += gridDim.x * blockDim.x) {
        fv4 v = W4[i];
        s += fabsf(v[0]) + fabsf(v[1]) + fabsf(v[2]) + fabsf(v[3]);
    }
    red[tid] = s;
    __syncthreads();
    #pragma unroll
    for (int off = 128; off > 0; off >>= 1) {
        if (tid < off) red[tid] += red[tid + off];
        __syncthreads();
    }
    if (tid == 0) part[blockIdx.x] = red[0];
}

// ---------------- pass 2: gamma -> scale = 1/(gamma+eps) --------------------
__global__ __launch_bounds__(256)
void bl_finalize_scale(const float* __restrict__ part, float* __restrict__ scale_out,
                       float inv_count) {
    __shared__ float red[256];
    const int tid = threadIdx.x;
    red[tid] = part[tid];           // RED_BLOCKS == 256 partials
    __syncthreads();
    #pragma unroll
    for (int off = 128; off > 0; off >>= 1) {
        if (tid < off) red[tid] += red[tid + off];
        __syncthreads();
    }
    if (tid == 0) {
        float gamma = red[0] * inv_count;
        scale_out[0] = 1.f / (gamma + QEPS);
    }
}

// ternary quantize one weight, result exact in f16
__device__ inline _Float16 bl_tern(float w, float s) {
    float q = __builtin_rintf(w * s);            // round-half-to-even, like jnp.round
    q = fminf(1.f, fmaxf(-1.f, q));
    return (_Float16)q;
}

// stage one 256x32 x-tile (f32->f16) and 128x32 W-tile (f32->ternary f16)
__device__ inline void bl_stage(const float* __restrict__ xrow,
                                const float* __restrict__ wrow,
                                _Float16* __restrict__ As,
                                _Float16* __restrict__ Bs,
                                int k0, int tid, int wRow, int wCol, float scale) {
    #pragma unroll
    for (int i = 0; i < 4; ++i) {
        fv4 a0 = *(const fv4*)(xrow + k0 + i * 8);
        fv4 a1 = *(const fv4*)(xrow + k0 + i * 8 + 4);
        h8 ha = { (_Float16)a0[0], (_Float16)a0[1], (_Float16)a0[2], (_Float16)a0[3],
                  (_Float16)a1[0], (_Float16)a1[1], (_Float16)a1[2], (_Float16)a1[3] };
        *(h8*)&As[tid * LDT + i * 8] = ha;
    }
    #pragma unroll
    for (int i = 0; i < 2; ++i) {
        fv4 w0 = *(const fv4*)(wrow + k0 + i * 8);
        fv4 w1 = *(const fv4*)(wrow + k0 + i * 8 + 4);
        h8 hw = { bl_tern(w0[0], scale), bl_tern(w0[1], scale),
                  bl_tern(w0[2], scale), bl_tern(w0[3], scale),
                  bl_tern(w1[0], scale), bl_tern(w1[1], scale),
                  bl_tern(w1[2], scale), bl_tern(w1[3], scale) };
        *(h8*)&Bs[wRow * LDT + wCol + i * 8] = hw;
    }
}

// ---------------- pass 3: WMMA GEMM (double-buffered LDS) -------------------
__global__ __launch_bounds__(256)
void bl_gemm_wmma(const float* __restrict__ x, const float* __restrict__ W,
                  const float* __restrict__ bias, const float* __restrict__ scale_p,
                  float* __restrict__ out) {
    __shared__ _Float16 As[2][BM * LDT];   // x tiles,  256 rows (m) x 32 k
    __shared__ _Float16 Bs[2][BN * LDT];   // Wq tiles, 128 rows (n) x 32 k

    const float scale = scale_p[0];

    const int tid   = threadIdx.x;
    const int lane  = tid & 31;
    const int wave  = tid >> 5;               // 0..7
    const int waveM = (wave >> 1) * 64;       // 4 waves along M
    const int waveN = (wave & 1) * 64;        // 2 waves along N
    const int hsel  = lane >> 4;              // 0: K 0-7/16-23, 1: K 8-15/24-31
    const int l16   = lane & 15;

    const int blockM = blockIdx.y * BM;
    const int blockN = blockIdx.x * BN;

    // staging: x -> 1 thread per row (32 floats); W -> 2 threads per row (16 each)
    const int wRow = tid >> 1;                // 0..127
    const int wCol = (tid & 1) * 16;          // 0 or 16

    const float* xrow = x + (size_t)(blockM + tid)  * KDIM;
    const float* wrow = W + (size_t)(blockN + wRow) * KDIM + wCol;

    v8f acc[4][4];
    #pragma unroll
    for (int mt = 0; mt < 4; ++mt)
        #pragma unroll
        for (int nt = 0; nt < 4; ++nt)
            acc[mt][nt] = (v8f)0.f;

    // prologue: stage tile 0 into buffer 0
    bl_stage(xrow, wrow, As[0], Bs[0], 0, tid, wRow, wCol, scale);
    __syncthreads();

    const int kb = hsel * 8;                  // per-lane K base within tile
    int buf = 0;

    for (int k0 = 0; k0 < KDIM; k0 += BK, buf ^= 1) {
        // stage next tile into the other buffer while this one computes
        if (k0 + BK < KDIM) {
            bl_stage(xrow, wrow, As[buf ^ 1], Bs[buf ^ 1], k0 + BK,
                     tid, wRow, wCol, scale);
            if (k0 + 2 * BK < KDIM) {         // prime L2 one tile further out
                __builtin_prefetch(xrow + k0 + 2 * BK, 0, 1);
                __builtin_prefetch(wrow + k0 + 2 * BK, 0, 1);
            }
        }

        // ---- fragment loads per the 16-bit A/B VGPR layout ----
        const _Float16* Ab = As[buf];
        const _Float16* Bb = Bs[buf];

        v16h bfrag[4];
        #pragma unroll
        for (int nt = 0; nt < 4; ++nt) {
            const int rn = waveN + nt * 16 + l16;
            h8 lo = *(const h8*)&Bb[rn * LDT + kb];
            h8 hi = *(const h8*)&Bb[rn * LDT + 16 + kb];
            bfrag[nt] = __builtin_shufflevector(lo, hi, 0,1,2,3,4,5,6,7,8,9,10,11,12,13,14,15);
        }

        #pragma unroll
        for (int mt = 0; mt < 4; ++mt) {
            const int rm = waveM + mt * 16 + l16;
            h8 lo = *(const h8*)&Ab[rm * LDT + kb];
            h8 hi = *(const h8*)&Ab[rm * LDT + 16 + kb];
            v16h afrag = __builtin_shufflevector(lo, hi, 0,1,2,3,4,5,6,7,8,9,10,11,12,13,14,15);

            #pragma unroll
            for (int nt = 0; nt < 4; ++nt) {
                acc[mt][nt] = __builtin_amdgcn_wmma_f32_16x16x32_f16(
                    false, afrag, false, bfrag[nt], (short)0, acc[mt][nt], false, false);
            }
        }
        __syncthreads();    // one barrier per K-step: next buffer is ready
    }

    // ---- epilogue: C/D layout (VGPR r, lane) -> M = r + 8*hsel, N = l16 ----
    #pragma unroll
    for (int nt = 0; nt < 4; ++nt) {
        const int gn = blockN + waveN + nt * 16 + l16;
        const float bv = bias[gn];
        #pragma unroll
        for (int mt = 0; mt < 4; ++mt) {
            const int gmBase = blockM + waveM + mt * 16 + hsel * 8;
            #pragma unroll
            for (int r = 0; r < 8; ++r) {
                out[(size_t)(gmBase + r) * NDIM + gn] = acc[mt][nt][r] + bv;
            }
        }
    }
}

// ---------------------------------------------------------------------------
extern "C" void kernel_launch(void* const* d_in, const int* in_sizes, int n_in,
                              void* d_out, int out_size, void* d_ws, size_t ws_size,
                              hipStream_t stream) {
    const float* x    = (const float*)d_in[0];   // [4,2048,4096]
    const float* W    = (const float*)d_in[1];   // [4096,4096]
    const float* bias = (const float*)d_in[2];   // [4096]
    float* out = (float*)d_out;
    float* wsf = (float*)d_ws;                   // [0..255] partials, [256] scale

    const int nW = NDIM * KDIM;                  // 16,777,216

    bl_abs_partial<<<RED_BLOCKS, 256, 0, stream>>>(W, wsf, nW / 4);
    bl_finalize_scale<<<1, 256, 0, stream>>>(wsf, wsf + RED_BLOCKS, 1.f / (float)nW);

    dim3 grid(NDIM / BN, MDIM / BM);             // (32, 32)
    bl_gemm_wmma<<<grid, 256, 0, stream>>>(x, W, bias, wsf + RED_BLOCKS, out);
}
